// StudentMoELayer_51453708206115
// MI455X (gfx1250) — matrix operand order repaired and checked by reading, not compiled, not run on hardware
//
#include <hip/hip_runtime.h>
#include <hip/hip_bf16.h>

// ---------------------------------------------------------------------------
// CDNA5 (gfx1250) MoE layer: RMSNorm router + top-2 dispatch + bf16 WMMA FFN.
// Async global->LDS staging (ASYNCcnt), double-buffered LDS, WMMA bf16 math.
// ---------------------------------------------------------------------------

typedef __attribute__((ext_vector_type(16))) __bf16 v16bf;
typedef __attribute__((ext_vector_type(8)))  float  v8f;

union Frag {
    uint4 q[2];
    v16bf v;
};

#define NUM_EXPERTS 8
#define TOP_K 2
#define SCALING 4.0f
#define RMS_EPS 1e-6f

// Round-to-nearest-even f32 -> bf16 (bit manipulation).
__device__ __forceinline__ unsigned short f2bf(float f) {
    unsigned int u = __float_as_uint(f);
    u += 0x7FFFu + ((u >> 16) & 1u);
    return (unsigned short)(u >> 16);
}

// --- CDNA5 async global->LDS copy, 16B per lane, tracked by ASYNCcnt --------
__device__ __forceinline__ void async_b128(unsigned lds_off, const void* gsrc) {
    asm volatile("global_load_async_to_lds_b128 %0, %1, off"
                 :: "v"(lds_off), "v"(gsrc) : "memory");
}
__device__ __forceinline__ void wait_async0() {
    asm volatile("s_wait_asynccnt 0x0" ::: "memory");
}
// Low 32 bits of a generic pointer to __shared__ = LDS byte offset.
__device__ __forceinline__ unsigned lds_off(const void* p) {
    return (unsigned)(unsigned long long)p;
}

// -------- WMMA fragment loaders (16-bit A 16x32 / B 32x16 layouts, wave32) ---
__device__ __forceinline__ v16bf load_frag_a(const unsigned short* s, int row0, int ld) {
    int lane = threadIdx.x & 31;
    int r  = row0 + (lane & 15);
    int kb = (lane >> 4) * 8;
    Frag f;
    f.q[0] = *(const uint4*)(s + r * ld + kb);
    f.q[1] = *(const uint4*)(s + r * ld + kb + 16);
    return f.v;
}
__device__ __forceinline__ v16bf load_frag_b(const unsigned short* s, int col0, int ld) {
    int lane = threadIdx.x & 31;
    int n  = col0 + (lane & 15);
    int kb = (lane >> 4) * 8;
    Frag f;
    f.q[0] = *(const uint4*)(s + n * ld + kb);
    f.q[1] = *(const uint4*)(s + n * ld + kb + 16);
    return f.v;
}

// ---------------------------------------------------------------------------
// Streaming f32 -> bf16 conversion (8 elements / thread)
// ---------------------------------------------------------------------------
__global__ __launch_bounds__(256) void cvt_f32_bf16(
    const float* __restrict__ in, unsigned short* __restrict__ out, long n)
{
    long i = ((long)blockIdx.x * 256 + threadIdx.x) * 8;
    if (i + 8 <= n) {
        float4 a = *(const float4*)(in + i);
        float4 b = *(const float4*)(in + i + 4);
        uint4 o;
        o.x = (unsigned)f2bf(a.x) | ((unsigned)f2bf(a.y) << 16);
        o.y = (unsigned)f2bf(a.z) | ((unsigned)f2bf(a.w) << 16);
        o.z = (unsigned)f2bf(b.x) | ((unsigned)f2bf(b.y) << 16);
        o.w = (unsigned)f2bf(b.z) | ((unsigned)f2bf(b.w) << 16);
        *(uint4*)(out + i) = o;
    }
}

// ---------------------------------------------------------------------------
// Router: RMSNorm -> logits -> softmax -> top-2 -> renormalized token weights.
// One block (256 threads = 8 waves) per token.
// ---------------------------------------------------------------------------
__global__ __launch_bounds__(256) void moe_router(
    const float* __restrict__ x,        // [T,H]
    const float* __restrict__ norm_w,   // [H]
    const float* __restrict__ rw,       // [E,H]
    float* __restrict__ tokw,           // [E,T] (pre-zeroed)
    float* __restrict__ importance,     // [E]   (pre-zeroed)
    int*   __restrict__ loadcnt,        // [E]   (pre-zeroed)
    int T, int H)
{
    const int t    = blockIdx.x;
    const int tid  = threadIdx.x;
    const int lane = tid & 31;
    const int wave = tid >> 5;
    const float* xt = x + (size_t)t * H;

    float ss = 0.0f;
    for (int k = tid; k < H; k += 256) {
        float v = xt[k];
        ss += v * v;
    }
    for (int o = 16; o > 0; o >>= 1) ss += __shfl_down(ss, o, 32);
    __shared__ float s_red[8];
    __shared__ float s_scale;
    if (lane == 0) s_red[wave] = ss;
    __syncthreads();
    if (tid == 0) {
        float tot = 0.0f;
        for (int w = 0; w < 8; ++w) tot += s_red[w];
        s_scale = rsqrtf(tot / (float)H + RMS_EPS);
    }
    __syncthreads();
    const float scale = s_scale;

    float p[NUM_EXPERTS];
#pragma unroll
    for (int e = 0; e < NUM_EXPERTS; ++e) p[e] = 0.0f;
    for (int k = tid; k < H; k += 256) {
        float xn = xt[k] * norm_w[k];
#pragma unroll
        for (int e = 0; e < NUM_EXPERTS; ++e) p[e] += xn * rw[(size_t)e * H + k];
    }
#pragma unroll
    for (int e = 0; e < NUM_EXPERTS; ++e)
        for (int o = 16; o > 0; o >>= 1) p[e] += __shfl_down(p[e], o, 32);

    __shared__ float s_pe[NUM_EXPERTS * 8];
    if (lane == 0) {
#pragma unroll
        for (int e = 0; e < NUM_EXPERTS; ++e) s_pe[e * 8 + wave] = p[e];
    }
    __syncthreads();

    if (tid == 0) {
        float logit[NUM_EXPERTS];
        float mx = -1e30f;
#pragma unroll
        for (int e = 0; e < NUM_EXPERTS; ++e) {
            float tot = 0.0f;
            for (int w = 0; w < 8; ++w) tot += s_pe[e * 8 + w];
            logit[e] = tot * scale;
            mx = fmaxf(mx, logit[e]);
        }
        float prob[NUM_EXPERTS];
        float sum = 0.0f;
#pragma unroll
        for (int e = 0; e < NUM_EXPERTS; ++e) { prob[e] = __expf(logit[e] - mx); sum += prob[e]; }
        float inv = 1.0f / sum;
#pragma unroll
        for (int e = 0; e < NUM_EXPERTS; ++e) {
            prob[e] *= inv;
            atomicAdd(&importance[e], prob[e]);
        }
        int i1 = 0;
#pragma unroll
        for (int e = 1; e < NUM_EXPERTS; ++e) if (prob[e] > prob[i1]) i1 = e;
        int i2 = (i1 == 0) ? 1 : 0;
#pragma unroll
        for (int e = 0; e < NUM_EXPERTS; ++e)
            if (e != i1 && prob[e] > prob[i2]) i2 = e;
        float w1 = prob[i1], w2 = prob[i2];
        float r = 1.0f / (w1 + w2);
        tokw[(size_t)i1 * T + t] = w1 * r;
        tokw[(size_t)i2 * T + t] = w2 * r;
        atomicAdd(&loadcnt[i1], 1);
        atomicAdd(&loadcnt[i2], 1);
    }
}

// ---------------------------------------------------------------------------
// Gate+Up fused GEMM (all-bf16 operands): h = silu(x@gT) * (x@uT), bf16 out.
// Block tile 128(M) x 64(N), BK=32; async double-buffered LDS staging.
// ---------------------------------------------------------------------------
#define BM 128
#define BN 64
#define BK 32

__global__ __launch_bounds__(256) void moe_gate_up(
    const unsigned short* __restrict__ x,   // [T,H] bf16
    const unsigned short* __restrict__ gw,  // [I,H] bf16 (expert slice)
    const unsigned short* __restrict__ uw,  // [I,H] bf16
    const float* __restrict__ tokw,         // [T] (this expert)
    unsigned short* __restrict__ hbuf,      // [T,I] bf16
    int T, int H, int I)
{
    const int tid  = threadIdx.x;
    const int lane = tid & 31;
    const int wave = tid >> 5;
    const int bm = blockIdx.y * BM;
    const int bn = blockIdx.x * BN;
    const int wm = (wave >> 1) * 32;
    const int wn = (wave & 1) * 32;

    __shared__ __align__(16) unsigned short sA[2][BM * BK];
    __shared__ __align__(16) unsigned short sG[2][BN * BK];
    __shared__ __align__(16) unsigned short sU[2][BN * BK];
    __shared__ int s_active;

    if (tid == 0) s_active = 0;
    __syncthreads();
    if (tid < BM && tokw[bm + tid] != 0.0f) s_active = 1;
    __syncthreads();
    if (!s_active) return;   // no token in this tile routed to this expert

    const int arow = tid >> 1, acol = (tid & 1) * 16;   // A: 32B / thread
    const int brow = tid >> 2, bcol = (tid & 3) * 8;    // B: 16B / thread

    // async stage of one K tile into buffer `b`
    auto stage = [&](int b, int kt) {
        const unsigned short* asrc = x  + (size_t)(bm + arow) * H + kt + acol;
        async_b128(lds_off(&sA[b][arow * BK + acol]),     asrc);
        async_b128(lds_off(&sA[b][arow * BK + acol + 8]), asrc + 8);
        async_b128(lds_off(&sG[b][brow * BK + bcol]),
                   gw + (size_t)(bn + brow) * H + kt + bcol);
        async_b128(lds_off(&sU[b][brow * BK + bcol]),
                   uw + (size_t)(bn + brow) * H + kt + bcol);
    };

    v8f ag[2][2], au[2][2];
#pragma unroll
    for (int i = 0; i < 2; ++i)
#pragma unroll
        for (int j = 0; j < 2; ++j) { ag[i][j] = (v8f)0.0f; au[i][j] = (v8f)0.0f; }

    stage(0, 0);
    wait_async0();
    __syncthreads();

    int cur = 0;
    for (int kt = 0; kt < H; kt += BK) {
        if (kt + BK < H) stage(cur ^ 1, kt + BK);   // fill next buffer async

        v16bf a0 = load_frag_a(sA[cur], wm,      BK);
        v16bf a1 = load_frag_a(sA[cur], wm + 16, BK);
        v16bf g0 = load_frag_b(sG[cur], wn,      BK);
        v16bf g1 = load_frag_b(sG[cur], wn + 16, BK);
        v16bf u0 = load_frag_b(sU[cur], wn,      BK);
        v16bf u1 = load_frag_b(sU[cur], wn + 16, BK);

        ag[0][0] = __builtin_amdgcn_wmma_f32_16x16x32_bf16(false, a0, false, g0, (short)0, ag[0][0], false, false);
        ag[0][1] = __builtin_amdgcn_wmma_f32_16x16x32_bf16(false, a0, false, g1, (short)0, ag[0][1], false, false);
        ag[1][0] = __builtin_amdgcn_wmma_f32_16x16x32_bf16(false, a1, false, g0, (short)0, ag[1][0], false, false);
        ag[1][1] = __builtin_amdgcn_wmma_f32_16x16x32_bf16(false, a1, false, g1, (short)0, ag[1][1], false, false);
        au[0][0] = __builtin_amdgcn_wmma_f32_16x16x32_bf16(false, a0, false, u0, (short)0, au[0][0], false, false);
        au[0][1] = __builtin_amdgcn_wmma_f32_16x16x32_bf16(false, a0, false, u1, (short)0, au[0][1], false, false);
        au[1][0] = __builtin_amdgcn_wmma_f32_16x16x32_bf16(false, a1, false, u0, (short)0, au[1][0], false, false);
        au[1][1] = __builtin_amdgcn_wmma_f32_16x16x32_bf16(false, a1, false, u1, (short)0, au[1][1], false, false);

        wait_async0();      // next tile resident in LDS
        __syncthreads();    // everyone done reading `cur`, everyone's fill done
        cur ^= 1;
    }

    // ---- epilogue: h = silu(g) * u, bf16 store ----
    const int mh = 8 * (lane >> 4);
    const int nl = lane & 15;
#pragma unroll
    for (int mi = 0; mi < 2; ++mi) {
#pragma unroll
        for (int ni = 0; ni < 2; ++ni) {
#pragma unroll
            for (int r = 0; r < 8; ++r) {
                float g = ag[mi][ni][r];
                float u = au[mi][ni][r];
                float v = (g / (1.0f + __expf(-g))) * u;
                int gm = bm + wm + mi * 16 + r + mh;
                int gn = bn + wn + ni * 16 + nl;
                hbuf[(size_t)gm * I + gn] = f2bf(v);
            }
        }
    }
}

// ---------------------------------------------------------------------------
// Down GEMM: out[t,h] += SCALING * w_e[t] * (h_bf16 @ downT)[t,h]
// ---------------------------------------------------------------------------
__global__ __launch_bounds__(256) void moe_down(
    const unsigned short* __restrict__ hbuf, // [T,I] bf16
    const unsigned short* __restrict__ dw,   // [H,I] bf16 (expert slice)
    const float* __restrict__ tokw,          // [T]
    float* __restrict__ out,                 // [T,H]
    int T, int H, int I)
{
    const int tid  = threadIdx.x;
    const int lane = tid & 31;
    const int wave = tid >> 5;
    const int bm = blockIdx.y * BM;
    const int bn = blockIdx.x * BN;
    const int wm = (wave >> 1) * 32;
    const int wn = (wave & 1) * 32;

    __shared__ __align__(16) unsigned short sA[2][BM * BK];
    __shared__ __align__(16) unsigned short sB[2][BN * BK];
    __shared__ float s_tokw[BM];
    __shared__ int s_active;

    if (tid == 0) s_active = 0;
    __syncthreads();
    if (tid < BM) {
        float w = tokw[bm + tid] * SCALING;
        s_tokw[tid] = w;
        if (w != 0.0f) s_active = 1;
    }
    __syncthreads();
    if (!s_active) return;

    const int arow = tid >> 1, acol = (tid & 1) * 16;
    const int brow = tid >> 2, bcol = (tid & 3) * 8;

    auto stage = [&](int b, int kt) {
        const unsigned short* asrc = hbuf + (size_t)(bm + arow) * I + kt + acol;
        async_b128(lds_off(&sA[b][arow * BK + acol]),     asrc);
        async_b128(lds_off(&sA[b][arow * BK + acol + 8]), asrc + 8);
        async_b128(lds_off(&sB[b][brow * BK + bcol]),
                   dw + (size_t)(bn + brow) * I + kt + bcol);
    };

    v8f acc[2][2];
#pragma unroll
    for (int i = 0; i < 2; ++i)
#pragma unroll
        for (int j = 0; j < 2; ++j) acc[i][j] = (v8f)0.0f;

    stage(0, 0);
    wait_async0();
    __syncthreads();

    int cur = 0;
    for (int kt = 0; kt < I; kt += BK) {
        if (kt + BK < I) stage(cur ^ 1, kt + BK);

        v16bf a0 = load_frag_a(sA[cur], wm,      BK);
        v16bf a1 = load_frag_a(sA[cur], wm + 16, BK);
        v16bf b0 = load_frag_b(sB[cur], wn,      BK);
        v16bf b1 = load_frag_b(sB[cur], wn + 16, BK);

        acc[0][0] = __builtin_amdgcn_wmma_f32_16x16x32_bf16(false, a0, false, b0, (short)0, acc[0][0], false, false);
        acc[0][1] = __builtin_amdgcn_wmma_f32_16x16x32_bf16(false, a0, false, b1, (short)0, acc[0][1], false, false);
        acc[1][0] = __builtin_amdgcn_wmma_f32_16x16x32_bf16(false, a1, false, b0, (short)0, acc[1][0], false, false);
        acc[1][1] = __builtin_amdgcn_wmma_f32_16x16x32_bf16(false, a1, false, b1, (short)0, acc[1][1], false, false);

        wait_async0();
        __syncthreads();
        cur ^= 1;
    }

    const int mh = 8 * (lane >> 4);
    const int nl = lane & 15;
#pragma unroll
    for (int mi = 0; mi < 2; ++mi) {
#pragma unroll
        for (int ni = 0; ni < 2; ++ni) {
#pragma unroll
            for (int r = 0; r < 8; ++r) {
                int ml = wm + mi * 16 + r + mh;
                int gm = bm + ml;
                int gn = bn + wn + ni * 16 + nl;
                float w = s_tokw[ml];
                if (w != 0.0f) {
                    out[(size_t)gm * H + gn] += acc[mi][ni][r] * w;
                }
            }
        }
    }
}

// ---------------------------------------------------------------------------
// Aux load-balancing loss finalize
// ---------------------------------------------------------------------------
__global__ void moe_aux(const float* __restrict__ importance,
                        const int* __restrict__ loadcnt,
                        float* __restrict__ out_aux, int T)
{
    if (threadIdx.x == 0 && blockIdx.x == 0) {
        float a = 0.0f;
        for (int e = 0; e < NUM_EXPERTS; ++e) {
            float imp  = importance[e] / (float)T;
            float load = (float)loadcnt[e] / (float)(T * TOP_K);
            a += imp * load;
        }
        out_aux[0] = a * (float)NUM_EXPERTS;
    }
}

// ---------------------------------------------------------------------------
extern "C" void kernel_launch(void* const* d_in, const int* in_sizes, int n_in,
                              void* d_out, int out_size, void* d_ws, size_t ws_size,
                              hipStream_t stream) {
    const float* x      = (const float*)d_in[0];  // [B,S,H]
    const float* norm_w = (const float*)d_in[1];  // [H]
    const float* rw     = (const float*)d_in[2];  // [E,H]
    const float* gate_w = (const float*)d_in[3];  // [E,I,H]
    const float* up_w   = (const float*)d_in[4];  // [E,I,H]
    const float* down_w = (const float*)d_in[5];  // [E,H,I]

    const int H = in_sizes[1];
    const int T = in_sizes[0] / H;
    const int E = in_sizes[2] / H;
    const int I = in_sizes[3] / (E * H);
    const long nX = (long)T * H;
    const long nW = (long)I * H;

    float* out = (float*)d_out;               // [T,H] + aux at [T*H]

    // workspace layout (bytes)
    char* ws = (char*)d_ws;
    float* tokw       = (float*)ws;                               // [E,T]
    float* importance = tokw + (size_t)E * T;                     // [E]
    int*   loadcnt    = (int*)(importance + NUM_EXPERTS);         // [E]
    unsigned short* xbf  = (unsigned short*)(ws + (1ull  << 20)); // [T,H]  32MB
    unsigned short* hbuf = (unsigned short*)(ws + (34ull << 20)); // [T,I]  16MB
    unsigned short* wgbf = (unsigned short*)(ws + (51ull << 20)); // [I,H]   4MB
    unsigned short* wubf = (unsigned short*)(ws + (56ull << 20)); // [I,H]   4MB
    unsigned short* wdbf = (unsigned short*)(ws + (61ull << 20)); // [H,I]   4MB

    hipMemsetAsync(d_ws, 0, ((size_t)E * T + 2 * NUM_EXPERTS) * sizeof(float), stream);
    hipMemsetAsync(d_out, 0, (size_t)T * H * sizeof(float), stream);

    moe_router<<<T, 256, 0, stream>>>(x, norm_w, rw, tokw, importance, loadcnt, T, H);

    // one-time activation conversion
    cvt_f32_bf16<<<(unsigned)(nX / (8 * 256)), 256, 0, stream>>>(x, xbf, nX);

    dim3 gridGU(I / BN, T / BM);
    dim3 gridDN(H / BN, T / BM);
    const unsigned cvtW = (unsigned)(nW / (8 * 256));
    for (int e = 0; e < NUM_EXPERTS; ++e) {
        cvt_f32_bf16<<<cvtW, 256, 0, stream>>>(gate_w + (size_t)e * nW, wgbf, nW);
        cvt_f32_bf16<<<cvtW, 256, 0, stream>>>(up_w   + (size_t)e * nW, wubf, nW);
        cvt_f32_bf16<<<cvtW, 256, 0, stream>>>(down_w + (size_t)e * nW, wdbf, nW);

        moe_gate_up<<<gridGU, 256, 0, stream>>>(
            xbf, wgbf, wubf, tokw + (size_t)e * T, hbuf, T, H, I);
        moe_down<<<gridDN, 256, 0, stream>>>(
            hbuf, wdbf, tokw + (size_t)e * T, out, T, H, I);
    }

    moe_aux<<<1, 64, 0, stream>>>(importance, loadcnt, out + (size_t)T * H, T);
}
